// PAINNEmbedding_57750130262471
// MI455X (gfx1250) — matrix operand order repaired and checked by reading, not compiled, not run on hardware
//
#include <hip/hip_runtime.h>
#include <hip/hip_bf16.h>
#include <math.h>

// ---------------------------------------------------------------------------
// PaiNN-style embedding for MI455X (gfx1250): bf16 WMMA GEMMs (fp32 accum),
// pre-packed B fragments (contiguous 32B/lane loads), L2-resident gathers,
// sorted-edge segment reduction (no atomics).
// ---------------------------------------------------------------------------

#define DIM   128
#define NC    128
#define NRAD  20
#define HID   128
#define OUT3  384   // DIM + 2*NC
#define NL    3
#define CUTOFF 5.0f
#define PI_F  3.14159265358979f

typedef __attribute__((ext_vector_type(16))) __bf16 v16bf;
typedef __attribute__((ext_vector_type(8)))  float  v8f;

static __device__ inline __bf16 f2bf(float f) {
    unsigned int u = __builtin_bit_cast(unsigned int, f);
    unsigned int r = (u + 0x7FFFu + ((u >> 16) & 1u)) >> 16;
    unsigned short s = (unsigned short)r;
    return __builtin_bit_cast(__bf16, s);
}

static __device__ inline float silu_f(float x) {
    return x / (1.0f + __expf(-x));
}

// ---- WMMA fragment handling (CDNA5 wave32 layouts, CDNA5 ISA) ---

// A: 16x32 bf16, row-major source, pre-offset to (row0, k0), row stride lda.
// lane: M = lane&15, g = lane>>4; VGPR0-3 hold K=g*8+0..7 (contiguous),
// VGPR4-7 hold K=16+g*8+0..7 (contiguous) -> two 16B loads per lane.
static __device__ inline v16bf load_frag_a(const __bf16* __restrict__ A,
                                           int lda, int lane) {
    int g = lane >> 4, m = lane & 15;
    const __bf16* p = A + (size_t)m * lda + g * 8;
    v16bf a;
#pragma unroll
    for (int i = 0; i < 8; ++i) { a[i] = p[i]; a[8 + i] = p[16 + i]; }
    return a;
}

// B fragments are PRE-PACKED: for col-tile ct and k-chunk kc, lane l holds 16
// contiguous bf16 at Wp[(((ct*KC + kc)*32 + l)*16]. One aligned 32B load.
static __device__ inline v16bf load_frag_b_packed(const __bf16* __restrict__ Wp,
                                                  int ct, int kc, int KC, int lane) {
    size_t base = ((((size_t)ct * KC + kc) * 32) + lane) << 4;
    return *(const v16bf*)(Wp + base);
}

static __device__ inline v8f wmma_bf16(v16bf a, v16bf b, v8f c) {
    return __builtin_amdgcn_wmma_f32_16x16x32_bf16(
        false, a, false, b, (short)0, c, false, false);
}

// ---------------------------------------------------------------------------
// Weight packing: W [K][Ncols] f32 row-major -> packed bf16 B-fragments.
// Element (kc, ct, lane, i) sources W[(kc*32 + (lane>>4)*8 + (i&7) + (i>>3)*16)*Ncols + ct*16 + (lane&15)]
// ---------------------------------------------------------------------------
__global__ void pack_b_kernel(const float* __restrict__ W,
                              __bf16* __restrict__ Wp, int K, int Ncols) {
    long total = (long)K * Ncols;
    long t = (long)blockIdx.x * 256 + threadIdx.x;
    if (t >= total) return;
    int KC = K / 32;
    int i  = (int)(t & 15);
    int l  = (int)((t >> 4) & 31);
    long q = t >> 9;
    int kc = (int)(q % KC);
    int ct = (int)(q / KC);
    int row = kc * 32 + (l >> 4) * 8 + (i & 7) + ((i >> 3) << 4);
    int col = ct * 16 + (l & 15);
    Wp[t] = f2bf(W[(size_t)row * Ncols + col]);
}

// ---------------------------------------------------------------------------
// Small elementwise / setup kernels
// ---------------------------------------------------------------------------

__global__ void f32_to_bf16_kernel(const float* __restrict__ src,
                                   __bf16* __restrict__ dst, long count) {
    long i = (long)blockIdx.x * blockDim.x + threadIdx.x;
    if (i < count) dst[i] = f2bf(src[i]);
}

// xi = Wsp[species] + bsp ; Vi = 0
__global__ void embed_kernel(const int* __restrict__ species,
                             const float* __restrict__ Wsp,
                             const float* __restrict__ bsp,
                             float* __restrict__ xi, float* __restrict__ Vi) {
    int n = blockIdx.x, c = threadIdx.x;
    int sp = species[n];
    xi[(size_t)n * DIM + c] = Wsp[(size_t)sp * DIM + c] + bsp[c];
#pragma unroll
    for (int k = 0; k < 3; ++k) Vi[(size_t)n * (3 * NC) + k * NC + c] = 0.0f;
}

// per-edge: dist, cosine switch, unit dir, 20 Bessel radial basis fns
__global__ void edge_geo_kernel(const float* __restrict__ vec,
                                float* __restrict__ eswitch,
                                float* __restrict__ edir,
                                float* __restrict__ erb, int E) {
    int e = blockIdx.x * 256 + threadIdx.x;
    if (e >= E) return;
    float x = vec[(size_t)e * 3 + 0];
    float y = vec[(size_t)e * 3 + 1];
    float z = vec[(size_t)e * 3 + 2];
    float d = sqrtf(x * x + y * y + z * z);
    float inv = 1.0f / d;
    edir[(size_t)e * 3 + 0] = x * inv;
    edir[(size_t)e * 3 + 1] = y * inv;
    edir[(size_t)e * 3 + 2] = z * inv;
    eswitch[e] = (d < CUTOFF) ? 0.5f * (__cosf(PI_F * d / CUTOFF) + 1.0f) : 0.0f;
    float pref = sqrtf(2.0f / CUTOFF) * inv;
    float ph = PI_F * d / CUTOFF;
#pragma unroll
    for (int j = 1; j <= NRAD; ++j)
        erb[(size_t)e * NRAD + (j - 1)] = pref * __sinf((float)j * ph);
}

// offsets[n] = lower_bound(edge_src, n); edge_src is sorted
__global__ void offsets_kernel(const int* __restrict__ edge_src,
                               int* __restrict__ offsets, int N, int E) {
    int n = blockIdx.x * 256 + threadIdx.x;
    if (n > N) return;
    int lo = 0, hi = E;
    while (lo < hi) {
        int mid = (lo + hi) >> 1;
        if (edge_src[mid] < n) lo = mid + 1; else hi = mid;
    }
    offsets[n] = lo;
}

// ---------------------------------------------------------------------------
// Fused 2-layer MLP: out = silu(X @ W1 + b1) @ W2 + b2
// X: [rows,K1] bf16; W1p/W2p packed B-fragments; out: [rows,384] f32.
// One 256-thread block (8 wave32) per 16 rows. Hidden staged in LDS as bf16.
// ---------------------------------------------------------------------------
template <int K1>
__global__ void mlp2_kernel(const __bf16* __restrict__ X,
                            const __bf16* __restrict__ W1p,
                            const float* __restrict__ b1,
                            const __bf16* __restrict__ W2p,
                            const float* __restrict__ b2,
                            float* __restrict__ Out) {
    __shared__ __bf16 hbuf[16 * HID];
    constexpr int KC1 = K1 / 32;
    int tid = threadIdx.x;
    int lane = tid & 31, w = tid >> 5;       // 8 waves
    int row0 = blockIdx.x * 16;
    const __bf16* Xb = X + (size_t)row0 * K1;
    int g = lane >> 4, n = lane & 15;

    // stage 1: wave w computes hidden columns [16w, 16w+16)
    {
        v8f acc = {};
#pragma unroll
        for (int kc = 0; kc < KC1; ++kc) {
            v16bf a = load_frag_a(Xb + kc * 32, K1, lane);
            v16bf b = load_frag_b_packed(W1p, w, kc, KC1, lane);
            acc = wmma_bf16(a, b, acc);
        }
        int col = w * 16 + n;
        float bias = b1[col];
#pragma unroll
        for (int v = 0; v < 8; ++v)
            hbuf[(g * 8 + v) * HID + col] = f2bf(silu_f(acc[v] + bias));
    }
    __syncthreads();

    // stage 2: wave w handles output col tiles w, w+8, w+16 of 384/16=24
#pragma unroll
    for (int t = 0; t < 3; ++t) {
        int ct = w + t * 8;
        v8f acc = {};
#pragma unroll
        for (int kc = 0; kc < HID / 32; ++kc) {
            v16bf a = load_frag_a(&hbuf[kc * 32], HID, lane);      // LDS reads
            v16bf b = load_frag_b_packed(W2p, ct, kc, HID / 32, lane);
            acc = wmma_bf16(a, b, acc);
        }
        int col = ct * 16 + n;
        float bias = b2[col];
#pragma unroll
        for (int v = 0; v < 8; ++v)
            Out[(size_t)(row0 + g * 8 + v) * OUT3 + col] = acc[v] + bias;
    }
}

// ---------------------------------------------------------------------------
// WMMA GEMM, packed B: D[rows,Ncols] = A[rows,K] @ B.  256-thread blocks:
// 8 waves cover 8 consecutive col tiles of one 16-row tile.
// ---------------------------------------------------------------------------
__global__ void gemm_bf16_kernel(const __bf16* __restrict__ A,
                                 const __bf16* __restrict__ Bp,
                                 float* __restrict__ D,
                                 int K, int Ncols) {
    int tid = threadIdx.x;
    int lane = tid & 31, w = tid >> 5;
    int rt = blockIdx.x;
    int ct = blockIdx.y * 8 + w;
    int KC = K / 32;
    const __bf16* Ab = A + (size_t)rt * 16 * K;
    v8f acc = {};
    for (int kc = 0; kc < KC; ++kc) {
        v16bf a = load_frag_a(Ab + kc * 32, K, lane);
        v16bf b = load_frag_b_packed(Bp, ct, kc, KC, lane);
        acc = wmma_bf16(a, b, acc);
    }
    int g = lane >> 4, n = lane & 15;
    int col = ct * 16 + n;
#pragma unroll
    for (int v = 0; v < 8; ++v)
        D[(size_t)(rt * 16 + g * 8 + v) * Ncols + col] = acc[v];
}

// ---------------------------------------------------------------------------
// Message + segment-sum: one block (128 threads) per node; edge_src sorted so
// each node owns a contiguous edge range -> deterministic, atomic-free.
// ---------------------------------------------------------------------------
__global__ void message_kernel(const int* __restrict__ offsets,
                               const int* __restrict__ edge_dst,
                               const float* __restrict__ eswitch,
                               const float* __restrict__ edir,
                               const float* __restrict__ erb,
                               const float* __restrict__ Wr,   // [20,384] layer slice
                               const float* __restrict__ br,   // [384] layer slice
                               const float* __restrict__ phi,  // [N,384]
                               const float* __restrict__ xi,   // [N,128]
                               const float* __restrict__ Vi,   // [N,3,128]
                               float* __restrict__ xmsg,       // [N,128]
                               __bf16* __restrict__ vmsg_bf,   // [N,3,128]
                               int layer) {
    __shared__ float WrS[NRAD * OUT3];
    int n = blockIdx.x, c = threadIdx.x;      // 128 threads
    for (int i = c; i < NRAD * OUT3; i += 128) WrS[i] = Wr[i];
    __syncthreads();

    int e0 = offsets[n], e1 = offsets[n + 1];
    float xacc = 0.0f, va0 = 0.0f, va1 = 0.0f, va2 = 0.0f;
    float bw0 = br[c], bw1 = br[DIM + c], bw2 = br[2 * DIM + c];

    for (int e = e0; e < e1; ++e) {
        int dst = edge_dst[e];
        float sw = eswitch[e];
        float w0 = bw0, w1 = bw1, w2 = bw2;
        const float* rbp = erb + (size_t)e * NRAD;
#pragma unroll
        for (int k = 0; k < NRAD; ++k) {
            float r = rbp[k];
            const float* wr = &WrS[k * OUT3];
            w0 = fmaf(r, wr[c], w0);
            w1 = fmaf(r, wr[DIM + c], w1);
            w2 = fmaf(r, wr[2 * DIM + c], w2);
        }
        w0 *= sw; w1 *= sw; w2 *= sw;
        const float* ph = phi + (size_t)dst * OUT3;
        float m0 = ph[c] * w0;             // dxij
        float m1 = ph[DIM + c] * w1;       // hvv
        float m2 = ph[2 * DIM + c] * w2;   // hvs
        xacc += m0;
        float d0 = edir[(size_t)e * 3 + 0];
        float d1 = edir[(size_t)e * 3 + 1];
        float d2 = edir[(size_t)e * 3 + 2];
        va0 = fmaf(d0, m2, va0);
        va1 = fmaf(d1, m2, va1);
        va2 = fmaf(d2, m2, va2);
        if (layer > 0) {
            const float* Vd = Vi + (size_t)dst * (3 * NC);
            va0 = fmaf(Vd[c], m1, va0);
            va1 = fmaf(Vd[NC + c], m1, va1);
            va2 = fmaf(Vd[2 * NC + c], m1, va2);
        }
    }
    xmsg[(size_t)n * DIM + c] = xi[(size_t)n * DIM + c] + xacc;
    const float* Vn = Vi + (size_t)n * (3 * NC);
    size_t vb = (size_t)n * (3 * NC);
    vmsg_bf[vb + c]          = f2bf(Vn[c] + va0);
    vmsg_bf[vb + NC + c]     = f2bf(Vn[NC + c] + va1);
    vmsg_bf[vb + 2 * NC + c] = f2bf(Vn[2 * NC + c] + va2);
}

// scals = sum_k u*v ; norms = sqrt(sum_k v^2 + 1e-16) ; build bf16 [xmsg|norms]
__global__ void post_uv_kernel(const float* __restrict__ uv,    // [N,3,256]
                               const float* __restrict__ xmsg,  // [N,128]
                               float* __restrict__ scals,       // [N,128]
                               __bf16* __restrict__ catin) {    // [N,256]
    int n = blockIdx.x, c = threadIdx.x;      // 128 threads
    const float* U = uv + (size_t)n * (3 * 2 * NC);
    float s = 0.0f, q = 0.0f;
#pragma unroll
    for (int k = 0; k < 3; ++k) {
        float u = U[k * 2 * NC + c];
        float v = U[k * 2 * NC + NC + c];
        s = fmaf(u, v, s);
        q = fmaf(v, v, q);
    }
    scals[(size_t)n * NC + c] = s;
    catin[(size_t)n * 2 * NC + c]      = f2bf(xmsg[(size_t)n * DIM + c]);
    catin[(size_t)n * 2 * NC + NC + c] = f2bf(sqrtf(q + 1e-16f));
}

// xi += ass + scals*asv ; Vi += u*avv
__global__ void final_update_kernel(const float* __restrict__ A,   // [N,384]
                                    const float* __restrict__ uv,  // [N,3,256]
                                    const float* __restrict__ scals,
                                    float* __restrict__ xi,
                                    float* __restrict__ Vi) {
    int n = blockIdx.x, c = threadIdx.x;
    const float* An = A + (size_t)n * OUT3;
    float ass = An[c], asv = An[DIM + c], avv = An[2 * DIM + c];
    xi[(size_t)n * DIM + c] += ass + scals[(size_t)n * NC + c] * asv;
    const float* U = uv + (size_t)n * (3 * 2 * NC);
    float* Vn = Vi + (size_t)n * (3 * NC);
#pragma unroll
    for (int k = 0; k < 3; ++k)
        Vn[k * NC + c] += U[k * 2 * NC + c] * avv;
}

// ---------------------------------------------------------------------------

static inline size_t alignUp256(size_t x) { return (x + 255) & ~(size_t)255; }

extern "C" void kernel_launch(void* const* d_in, const int* in_sizes, int n_in,
                              void* d_out, int out_size, void* d_ws, size_t ws_size,
                              hipStream_t stream) {
    (void)n_in; (void)out_size; (void)ws_size;
    const int N = in_sizes[0];
    const int E = in_sizes[1] / 3;

    const int*   species  = (const int*)  d_in[0];
    const float* vec      = (const float*)d_in[1];
    const int*   edge_src = (const int*)  d_in[2];
    const int*   edge_dst = (const int*)  d_in[3];
    const float* Wsp      = (const float*)d_in[4];
    const float* bsp      = (const float*)d_in[5];
    const float* Wm1      = (const float*)d_in[6];
    const float* bm1      = (const float*)d_in[7];
    const float* Wm2      = (const float*)d_in[8];
    const float* bm2      = (const float*)d_in[9];
    const float* Wr       = (const float*)d_in[10];
    const float* br       = (const float*)d_in[11];
    const float* Wuv      = (const float*)d_in[12];
    const float* Wu1      = (const float*)d_in[13];
    const float* bu1      = (const float*)d_in[14];
    const float* Wu2      = (const float*)d_in[15];
    const float* bu2      = (const float*)d_in[16];

    float* xi = (float*)d_out;                    // [N,128]
    float* Vi = (float*)d_out + (size_t)N * DIM;  // [N,3,128]

    // ---- workspace carve ----
    char* w = (char*)d_ws;
    auto carve = [&](size_t bytes) { char* p = w; w += alignUp256(bytes); return p; };
    float*  eswitch = (float*) carve((size_t)E * 4);
    float*  edir    = (float*) carve((size_t)E * 3 * 4);
    float*  erb     = (float*) carve((size_t)E * NRAD * 4);
    int*    offsets = (int*)   carve((size_t)(N + 1) * 4);
    __bf16* xi_bf   = (__bf16*)carve((size_t)N * DIM * 2);
    float*  phi     = (float*) carve((size_t)N * OUT3 * 4);
    float*  xmsg    = (float*) carve((size_t)N * DIM * 4);
    __bf16* vmsg_bf = (__bf16*)carve((size_t)N * 3 * NC * 2);
    float*  uv      = (float*) carve((size_t)N * 3 * 2 * NC * 4);
    float*  scals   = (float*) carve((size_t)N * NC * 4);
    __bf16* catin   = (__bf16*)carve((size_t)N * 2 * NC * 2);
    float*  Abuf    = (float*) carve((size_t)N * OUT3 * 4);
    __bf16* Wm1p    = (__bf16*)carve((size_t)NL * DIM * HID * 2);
    __bf16* Wm2p    = (__bf16*)carve((size_t)NL * HID * OUT3 * 2);
    __bf16* Wuvp    = (__bf16*)carve((size_t)NL * NC * 2 * NC * 2);
    __bf16* Wu1p    = (__bf16*)carve((size_t)NL * (DIM + NC) * HID * 2);
    __bf16* Wu2p    = (__bf16*)carve((size_t)NL * HID * OUT3 * 2);

    // ---- one-time weight packing into WMMA B-fragment order ----
    auto packB = [&](const float* s, __bf16* d, int K, int Ncols) {
        for (int l = 0; l < NL; ++l) {
            long cnt = (long)K * Ncols;
            pack_b_kernel<<<(unsigned)((cnt + 255) / 256), 256, 0, stream>>>(
                s + (size_t)l * K * Ncols, d + (size_t)l * K * Ncols, K, Ncols);
        }
    };
    packB(Wm1, Wm1p, DIM, HID);
    packB(Wm2, Wm2p, HID, OUT3);
    packB(Wuv, Wuvp, NC, 2 * NC);
    packB(Wu1, Wu1p, DIM + NC, HID);
    packB(Wu2, Wu2p, HID, OUT3);

    // ---- graph preprocessing ----
    embed_kernel<<<N, DIM, 0, stream>>>(species, Wsp, bsp, xi, Vi);
    edge_geo_kernel<<<(E + 255) / 256, 256, 0, stream>>>(vec, eswitch, edir, erb, E);
    offsets_kernel<<<(N + 256) / 256, 256, 0, stream>>>(edge_src, offsets, N, E);

    const int rowTiles = N / 16;        // 20000/16 = 1250 (exact)
    const int uvTiles  = (3 * N) / 16;  // 3750 (exact)

    for (int l = 0; l < NL; ++l) {
        // phi = silu(xi@Wm1+bm1)@Wm2+bm2
        f32_to_bf16_kernel<<<(unsigned)(((long)N * DIM + 255) / 256), 256, 0, stream>>>(
            xi, xi_bf, (long)N * DIM);
        mlp2_kernel<DIM><<<rowTiles, 256, 0, stream>>>(
            xi_bf, Wm1p + (size_t)l * DIM * HID, bm1 + (size_t)l * HID,
            Wm2p + (size_t)l * HID * OUT3, bm2 + (size_t)l * OUT3, phi);

        // edge messages + sorted-segment reduction
        message_kernel<<<N, DIM, 0, stream>>>(
            offsets, edge_dst, eswitch, edir, erb,
            Wr + (size_t)l * NRAD * OUT3, br + (size_t)l * OUT3,
            phi, xi, Vi, xmsg, vmsg_bf, l);

        // uv = v_message @ Wuv   ([3N,128]@[128,256]) ; 16 col tiles = 2 blocks of 8 waves
        gemm_bf16_kernel<<<dim3(uvTiles, (2 * NC) / (16 * 8)), 256, 0, stream>>>(
            vmsg_bf, Wuvp + (size_t)l * NC * 2 * NC, uv, NC, 2 * NC);

        // scals / norms / cat input
        post_uv_kernel<<<N, NC, 0, stream>>>(uv, xmsg, scals, catin);

        // A = silu(cat@Wu1+bu1)@Wu2+bu2
        mlp2_kernel<2 * NC><<<rowTiles, 256, 0, stream>>>(
            catin, Wu1p + (size_t)l * (DIM + NC) * HID, bu1 + (size_t)l * HID,
            Wu2p + (size_t)l * HID * OUT3, bu2 + (size_t)l * OUT3, Abuf);

        // xi += ass + scals*asv ; Vi += u*avv
        final_update_kernel<<<N, NC, 0, stream>>>(Abuf, uv, scals, xi, Vi);
    }
}
